// SwinTransformerBlock3D_31937376813417
// MI455X (gfx1250) — matrix operand order repaired and checked by reading, not compiled, not run on hardware
//
#include <hip/hip_runtime.h>
#include <math.h>

// ---------- types ----------
typedef __bf16 bf16_t;
typedef __attribute__((ext_vector_type(16))) __bf16 v16bf;
typedef __attribute__((ext_vector_type(8)))  __bf16 v8bf;
typedef __attribute__((ext_vector_type(8)))  float  v8f;
typedef __attribute__((ext_vector_type(4))) unsigned int u32x4;
typedef __attribute__((ext_vector_type(8))) int          i32x8;
typedef __attribute__((ext_vector_type(4))) int          i32x4;

// ---------- constants ----------
// B=2, D=16, H=56, W=56, C=512, window 2x7x7 -> N=98, 8x8x8 windows/batch
#define MTOK   100352           // total token rows = 1024 windows * 98
#define NWIN   1024
#define CCH    512
#define MLPD   2048

// ---------- Tensor Data Mover: 2D tile (bf16) global -> LDS ----------
// Pads 4 DWORDs after every 16 DWORDs stored => LDS row stride = 32 bf16 + 8 = 40.
// tile_d0 elems per row (must make a 4B-multiple row), tile_d1 rows,
// stride_elems = row stride of the source tensor in elements.
__device__ __forceinline__ void tdm_load_2d_bf16(unsigned lds_byte_addr, const void* gptr,
                                                 unsigned tile_d0, unsigned tile_d1,
                                                 unsigned long long stride_elems,
                                                 unsigned tensor_d0) {
    unsigned long long ga = (unsigned long long)gptr;
    u32x4 g0 = {};
    g0[0] = 1u;                                            // count=1, load, user mode
    g0[1] = lds_byte_addr;                                 // LDS dest (bytes)
    g0[2] = (unsigned)(ga & 0xffffffffu);                  // global addr [31:0]
    g0[3] = (unsigned)((ga >> 32) & 0x01ffffffu) | (2u << 30); // addr[56:32] | type=2
    const unsigned tensor_d1 = 1u << 30;                   // effectively no row OOB
    i32x8 g1 = {};
    g1[0] = (int)((1u << 16)      // data_size = 2 bytes
                | (1u << 20)      // pad_enable
                | (3u << 22)      // pad_interval: every 16 DWORDs
                | (3u << 25));    // pad_amount: 4 DWORDs
    g1[1] = (int)((tensor_d0 & 0xffffu) << 16);            // barrier_addr=0 | td0[15:0]
    g1[2] = (int)((tensor_d0 >> 16) | ((tensor_d1 & 0xffffu) << 16));
    g1[3] = (int)((tensor_d1 >> 16) | (tile_d0 << 16));
    g1[4] = (int)(tile_d1 & 0xffffu);                      // tile_dim1 | tile_dim2=0
    g1[5] = (int)(stride_elems & 0xffffffffu);             // dim0 stride [31:0]
    g1[6] = (int)((stride_elems >> 32) & 0xffffu);         // stride[47:32] | dim1_stride lo = 0
    g1[7] = 0;
    i32x4 z4 = {};
#if __has_include(<hip/amd_detail/amd_gfx1250_TDM.h>)
    i32x8 z8 = {};
    __builtin_amdgcn_tensor_load_to_lds(g0, g1, z4, z4, z8, 0);
#else
    __builtin_amdgcn_tensor_load_to_lds(g0, g1, z4, z4, 0);
#endif
}

// window-major row -> natural row (window reverse / partition mapping)
__device__ __forceinline__ size_t win_to_nat(int r) {
    int win = r / 98, t = r - win * 98;
    int b  = win >> 9, rem = win & 511;
    int dI = rem >> 6, hI = (rem >> 3) & 7, wI = rem & 7;
    int dw = t / 49, t2 = t - dw * 49, hw = t2 / 7, ww = t2 - hw * 7;
    return (size_t)b * (16 * 56 * 56)
         + (size_t)(dI * 2 + dw) * (56 * 56)
         + (size_t)(hI * 7 + hw) * 56
         + (size_t)(wI * 7 + ww);
}

// ---------- weight transpose + bf16 cast: wt[n*K+k] = w[k*N+n] ----------
__global__ void tcast_kernel(const float* __restrict__ w, bf16_t* __restrict__ wt,
                             int K, int N) {
    int idx = blockIdx.x * 256 + threadIdx.x;
    if (idx >= K * N) return;
    int k = idx / N, n = idx - k * N;
    wt[(size_t)n * K + k] = (bf16_t)w[idx];
}

// ---------- LayerNorm (optionally permuting rows window-major) -> bf16 ----------
__global__ __launch_bounds__(256)
void ln_kernel(const float* __restrict__ x, const float* __restrict__ g,
               const float* __restrict__ b, bf16_t* __restrict__ out, int permute) {
    __shared__ float red[256];
    __shared__ float red2[256];
    int row = blockIdx.x;
    size_t src = permute ? win_to_nat(row) : (size_t)row;
    const float* xr = x + src * CCH;
    int t = threadIdx.x;
    float v0 = xr[t], v1 = xr[t + 256];
    red[t]  = v0 + v1;
    red2[t] = v0 * v0 + v1 * v1;
    __syncthreads();
    for (int s = 128; s > 0; s >>= 1) {
        if (t < s) { red[t] += red[t + s]; red2[t] += red2[t + s]; }
        __syncthreads();
    }
    float mean = red[0] * (1.0f / CCH);
    float var  = red2[0] * (1.0f / CCH) - mean * mean;
    float rstd = rsqrtf(var + 1e-5f);
    bf16_t* orow = out + (size_t)row * CCH;
    orow[t]       = (bf16_t)((v0 - mean) * rstd * g[t]       + b[t]);
    orow[t + 256] = (bf16_t)((v1 - mean) * rstd * g[t + 256] + b[t + 256]);
}

// ---------- epilogue functors ----------
struct EpQKV {  // bias, store bf16 row-major ld=1536
    const float* bias; bf16_t* out;
    __device__ void operator()(int r, int c, float v) const {
        out[(size_t)r * (3 * CCH) + c] = (bf16_t)(v + bias[c]);
    }
};
struct EpProj { // bias + window-reverse into natural-layout fp32 xr
    const float* bias; float* xr;
    __device__ void operator()(int r, int c, float v) const {
        xr[win_to_nat(r) * CCH + c] = v + bias[c];
    }
};
struct EpGelu { // bias + tanh-gelu -> bf16 ld=2048
    const float* bias; bf16_t* out;
    __device__ void operator()(int r, int c, float v) const {
        float xv = v + bias[c];
        float gv = 0.5f * xv * (1.0f + tanhf(0.7978845608f * (xv + 0.044715f * xv * xv * xv)));
        out[(size_t)r * MLPD + c] = (bf16_t)gv;
    }
};
struct EpResid { // bias + residual add, fp32 output
    const float* bias; const float* xr; float* y;
    __device__ void operator()(int r, int c, float v) const {
        size_t i = (size_t)r * CCH + c;
        y[i] = xr[i] + v + bias[c];
    }
};

// ---------- bf16 WMMA GEMM: 128x128 tile, 8 waves, k-step 32 ----------
// Double-buffered LDS tiles filled by the Tensor Data Mover (wave 0 issues,
// TENSORcnt tracks, workgroup barrier publishes) so DMA overlaps WMMA.
// A: M x K row-major bf16 ; Bt: N x K row-major bf16 (i.e. B transposed)
template <class Ep>
__global__ __launch_bounds__(256)
void gemm_wmma_bf16(const bf16_t* __restrict__ A, const bf16_t* __restrict__ Bt,
                    int K, Ep ep) {
    constexpr int LDT = 40;                 // padded LDS stride (32 data + 8 pad)
    __shared__ bf16_t sA[2][128 * LDT];
    __shared__ bf16_t sB[2][128 * LDT];

    const int tid  = threadIdx.x;
    const int lane = tid & 31;
    const int wave = tid >> 5;
    const int wm = (wave & 3) * 32;         // wave row offset within tile
    const int wn = (wave >> 2) * 64;        // wave col offset within tile
    const size_t rowBase = (size_t)blockIdx.x * 128;
    const size_t colBase = (size_t)blockIdx.y * 128;

    const int hi = lane >> 4;
    const int lo = lane & 15;
    const int kbA = hi * 8;                 // A frag: k halves at kbA, kbA+16
    const int kbB = hi * 8;                 // B frag: k halves at kbB, kbB+16

    const bf16_t* Ab = A  + rowBase * K;
    const bf16_t* Bb = Bt + colBase * K;
    const bool issuer = (wave == 0);

    v8f zero = {};
    v8f acc[2][4];
#pragma unroll
    for (int i = 0; i < 2; ++i)
#pragma unroll
        for (int j = 0; j < 4; ++j) acc[i][j] = zero;

    if (issuer) {   // prime buffer 0
        tdm_load_2d_bf16((unsigned)(size_t)&sA[0][0], Ab, 32, 128, (unsigned long long)K, (unsigned)K);
        tdm_load_2d_bf16((unsigned)(size_t)&sB[0][0], Bb, 32, 128, (unsigned long long)K, (unsigned)K);
    }

    const int nk = K >> 5;
    for (int kk = 0; kk < nk; ++kk) {
        const int cur = kk & 1;
        if (issuer) __builtin_amdgcn_s_wait_tensorcnt(0);
        __syncthreads();                         // buf[cur] visible to all waves
        if (issuer && (kk + 1) < nk) {           // prefetch next slab into buf[1-cur]
            const int k1 = (kk + 1) << 5;
            tdm_load_2d_bf16((unsigned)(size_t)&sA[1 - cur][0], Ab + k1, 32, 128,
                             (unsigned long long)K, (unsigned)K);
            tdm_load_2d_bf16((unsigned)(size_t)&sB[1 - cur][0], Bb + k1, 32, 128,
                             (unsigned long long)K, (unsigned)K);
        }

        v16bf af[2], bfr[4];
#pragma unroll
        for (int i = 0; i < 2; ++i) {
            const bf16_t* p = &sA[cur][(wm + i * 16 + lo) * LDT + kbA];
            ((v8bf*)&af[i])[0] = *(const v8bf*)p;
            ((v8bf*)&af[i])[1] = *(const v8bf*)(p + 16);
        }
#pragma unroll
        for (int j = 0; j < 4; ++j) {
            const bf16_t* p = &sB[cur][(wn + j * 16 + lo) * LDT + kbB];
            ((v8bf*)&bfr[j])[0] = *(const v8bf*)p;
            ((v8bf*)&bfr[j])[1] = *(const v8bf*)(p + 16);
        }
#pragma unroll
        for (int i = 0; i < 2; ++i)
#pragma unroll
            for (int j = 0; j < 4; ++j)
                acc[i][j] = __builtin_amdgcn_wmma_f32_16x16x32_bf16(
                    false, af[i], false, bfr[j], (short)0, acc[i][j], false, false);
        __syncthreads();                         // all reads of buf[cur] done
    }

#pragma unroll
    for (int i = 0; i < 2; ++i)
#pragma unroll
        for (int j = 0; j < 4; ++j)
#pragma unroll
            for (int r = 0; r < 8; ++r)
                ep((int)(rowBase + wm + i * 16 + hi * 8 + r),
                   (int)(colBase + wn + j * 16 + lo), acc[i][j][r]);
}

// ---------- attention: one block per (window, head), 4 waves ----------
// qkv: (win*98+tok) x 1536 bf16 (q|k|v).  o: (win*98+tok) x 512 bf16.
// Q,K tiles (98x32, src stride 1536) fetched by TDM while other waves
// transpose V into LDS and zero the padding rows.
__global__ __launch_bounds__(128)
void attn_kernel(const bf16_t* __restrict__ qkv, bf16_t* __restrict__ o) {
    const int win  = blockIdx.x >> 4;
    const int head = blockIdx.x & 15;
    const int tid  = threadIdx.x;
    const int lane = tid & 31;
    const int wave = tid >> 5;
    const int hi = lane >> 4, lo = lane & 15;

    extern __shared__ char smem[];
    bf16_t* sQ  = (bf16_t*)smem;            // 112 x 40
    bf16_t* sK  = sQ + 112 * 40;            // 112 x 40
    bf16_t* sVT = sK + 112 * 40;            // 32  x 136 (V transposed, col=token)
    float*  sS  = (float*)(sVT + 32 * 136); // 112 x 132 fp32 scores
    // P (bf16, stride 264 elems) aliases S row-for-row: safe, each thread owns its row.
    bf16_t* sP  = (bf16_t*)sS;
    const int LDP = 264;

    const size_t qbase = (size_t)win * 98 * 1536 + (size_t)head * 32;

    // TDM: rows 0..97 of Q and K (LDS stride 40 via pad), DMA overlaps V work below
    if (wave == 0) {
        tdm_load_2d_bf16((unsigned)(size_t)sQ, qkv + qbase,       32, 98, 1536ull, 1536u);
        tdm_load_2d_bf16((unsigned)(size_t)sK, qkv + qbase + 512, 32, 98, 1536ull, 1536u);
    }
    // zero padding rows 98..111 of Q,K (disjoint from TDM's rows 0..97)
    for (int i = tid; i < 14 * 40; i += 128) {
        sQ[98 * 40 + i] = (bf16_t)0.0f;
        sK[98 * 40 + i] = (bf16_t)0.0f;
    }
    // load V transposed: sVT[hd][tok], zero tokens 98..127
    for (int i = tid; i < 4096; i += 128) {
        int t = i >> 5, hd = i & 31;
        bf16_t v = (bf16_t)0.0f;
        if (t < 98) v = qkv[qbase + (size_t)t * 1536 + 1024 + hd];
        sVT[hd * 136 + t] = v;
    }
    if (wave == 0) __builtin_amdgcn_s_wait_tensorcnt(0);
    __syncthreads();

    // S = Q @ K^T : 7x7 tiles, HD=32 == one wmma k-step
    for (int t = wave; t < 49; t += 4) {
        int mt = t / 7, nt = t - mt * 7;
        v16bf aQ, bK;
        const bf16_t* pa = &sQ[(mt * 16 + lo) * 40 + hi * 8];
        ((v8bf*)&aQ)[0] = *(const v8bf*)pa;
        ((v8bf*)&aQ)[1] = *(const v8bf*)(pa + 16);
        const bf16_t* pb = &sK[(nt * 16 + lo) * 40 + hi * 8];
        ((v8bf*)&bK)[0] = *(const v8bf*)pb;
        ((v8bf*)&bK)[1] = *(const v8bf*)(pb + 16);
        v8f s = {};
        s = __builtin_amdgcn_wmma_f32_16x16x32_bf16(false, aQ, false, bK,
                                                    (short)0, s, false, false);
#pragma unroll
        for (int r = 0; r < 8; ++r)
            sS[(mt * 16 + r + hi * 8) * 132 + nt * 16 + lo] = s[r];
    }
    __syncthreads();

    // softmax over valid 98 columns; write P bf16 in place (rows 98..111 -> 0)
    if (tid < 112) {
        float*  srow = sS + tid * 132;
        bf16_t* prow = sP + tid * LDP;
        if (tid < 98) {
            const float sc = 0.17677669529663689f; // 1/sqrt(32)
            float mx = -1e30f;
            for (int j = 0; j < 98; ++j) mx = fmaxf(mx, srow[j]);
            float sum = 0.0f;
            for (int j = 0; j < 98; ++j) { float e = __expf(sc * (srow[j] - mx)); srow[j] = e; sum += e; }
            float inv = 1.0f / sum;
            for (int j = 0; j < 98; ++j)  prow[j] = (bf16_t)(srow[j] * inv); // read j before clobbering j/2
            for (int j = 98; j < 128; ++j) prow[j] = (bf16_t)0.0f;
        } else {
            for (int j = 0; j < 128; ++j)  prow[j] = (bf16_t)0.0f;
        }
    }
    __syncthreads();

    // O = P @ V : 7 m-tiles x 2 n-tiles, K=128 (4 k-steps)
    for (int t = wave; t < 14; t += 4) {
        int mt = t >> 1, nt = t & 1;
        v8f oacc = {};
#pragma unroll
        for (int ks = 0; ks < 4; ++ks) {
            v16bf aP, bV;
            const bf16_t* pa = &sP[(mt * 16 + lo) * LDP + ks * 32 + hi * 8];
            ((v8bf*)&aP)[0] = *(const v8bf*)pa;
            ((v8bf*)&aP)[1] = *(const v8bf*)(pa + 16);
            const bf16_t* pb = &sVT[(nt * 16 + lo) * 136 + ks * 32 + hi * 8];
            ((v8bf*)&bV)[0] = *(const v8bf*)pb;
            ((v8bf*)&bV)[1] = *(const v8bf*)(pb + 16);
            oacc = __builtin_amdgcn_wmma_f32_16x16x32_bf16(false, aP, false, bV,
                                                           (short)0, oacc, false, false);
        }
#pragma unroll
        for (int r = 0; r < 8; ++r) {
            int tok = mt * 16 + r + hi * 8;
            if (tok < 98)
                o[((size_t)(win * 98 + tok)) * CCH + head * 32 + nt * 16 + lo] = (bf16_t)oacc[r];
        }
    }
}

// ---------- host launcher ----------
extern "C" void kernel_launch(void* const* d_in, const int* in_sizes, int n_in,
                              void* d_out, int out_size, void* d_ws, size_t ws_size,
                              hipStream_t stream) {
    const float* x     = (const float*)d_in[0];
    const float* g1    = (const float*)d_in[1];
    const float* b1    = (const float*)d_in[2];
    const float* wqkv  = (const float*)d_in[3];
    const float* bqkv  = (const float*)d_in[4];
    const float* wproj = (const float*)d_in[5];
    const float* bproj = (const float*)d_in[6];
    const float* g2    = (const float*)d_in[7];
    const float* b2    = (const float*)d_in[8];
    const float* w1    = (const float*)d_in[9];
    const float* bm1   = (const float*)d_in[10];
    const float* w2    = (const float*)d_in[11];
    const float* bm2   = (const float*)d_in[12];
    float* y = (float*)d_out;

    const size_t M = MTOK;
    char* ws = (char*)d_ws;
    // layout (bytes): h[M*1024] | qkv[M*3072] | owin[M*1024] | xr[M*2048] | weightsT
    bf16_t* h    = (bf16_t*)ws;
    bf16_t* qkvb = (bf16_t*)(ws + M * 1024);
    bf16_t* owin = (bf16_t*)(ws + M * 1024 + M * 3072);
    float*  xr   = (float*) (ws + M * 1024 + M * 4096);
    char*   wsp  =          ws + M * 1024 + M * 4096 + M * 2048;
    bf16_t* wqkvT  = (bf16_t*)wsp;            // 1536 x 512
    bf16_t* wprojT = wqkvT  + 1536 * 512;     // 512  x 512
    bf16_t* w1T    = wprojT + 512 * 512;      // 2048 x 512
    bf16_t* w2T    = w1T    + 2048 * 512;     // 512  x 2048
    bf16_t* a1 = qkvb;  // M x 2048, reuses qkv+owin region after proj consumes it
    bf16_t* h2 = h;     // reuses h after QKV GEMM consumes it

    // weights -> bf16 transposed
    tcast_kernel<<<(512 * 1536 + 255) / 256, 256, 0, stream>>>(wqkv,  wqkvT,  512, 1536);
    tcast_kernel<<<(512 * 512  + 255) / 256, 256, 0, stream>>>(wproj, wprojT, 512, 512);
    tcast_kernel<<<(512 * 2048 + 255) / 256, 256, 0, stream>>>(w1,    w1T,    512, 2048);
    tcast_kernel<<<(2048 * 512 + 255) / 256, 256, 0, stream>>>(w2,    w2T,    2048, 512);

    // window partition + LN1 -> h (bf16, window-major)
    ln_kernel<<<(unsigned)M, 256, 0, stream>>>(x, g1, b1, h, 1);

    // QKV projection
    { EpQKV ep{bqkv, qkvb};
      gemm_wmma_bf16<<<dim3(784, 12), 256, 0, stream>>>(h, wqkvT, 512, ep); }

    // attention per (window, head); ~84 KB dynamic LDS
    {
        size_t smem = (size_t)(112 * 40 + 112 * 40 + 32 * 136) * sizeof(bf16_t)
                    + (size_t)(112 * 132) * sizeof(float);
        attn_kernel<<<NWIN * 16, 128, smem, stream>>>(qkvb, owin);
    }

    // output projection + window reverse -> xr (fp32 natural layout)
    { EpProj ep{bproj, xr};
      gemm_wmma_bf16<<<dim3(784, 4), 256, 0, stream>>>(owin, wprojT, 512, ep); }

    // LN2 -> h2 (bf16)
    ln_kernel<<<(unsigned)M, 256, 0, stream>>>(xr, g2, b2, h2, 0);

    // MLP fc1 + GELU -> a1 (bf16)
    { EpGelu ep{bm1, a1};
      gemm_wmma_bf16<<<dim3(784, 16), 256, 0, stream>>>(h2, w1T, 512, ep); }

    // MLP fc2 + residual -> y (fp32)
    { EpResid ep{bm2, xr, y};
      gemm_wmma_bf16<<<dim3(784, 4), 256, 0, stream>>>(a1, w2T, 2048, ep); }
}
